// HinSAGENodeClassifier_19413252177996
// MI455X (gfx1250) — matrix operand order repaired and checked by reading, compiled, not run on hardware
//
#include <hip/hip_runtime.h>

// ---------------- problem constants (match reference) ----------------
#define NU   50000      // users
#define NPo  20000      // posts
#define IND  256        // input feature dim
#define EFD  64         // edge feature dim
#define HD   128        // hidden dim
#define CD   4          // classes
#define LL   2          // layers
#define EPu  200000     // publish edges
#define ECo  800000     // comment edges
#define EUc  300000     // user-comment-user edges

typedef __attribute__((ext_vector_type(16))) _Float16 v16h;
typedef __attribute__((ext_vector_type(8)))  _Float16 v8h;
typedef __attribute__((ext_vector_type(8)))  float    v8f;
typedef __attribute__((ext_vector_type(4)))  unsigned int v4u;
typedef __attribute__((ext_vector_type(8)))  int v8i;
typedef __attribute__((ext_vector_type(4)))  int v4i;

#if defined(__gfx1250__) && __has_builtin(__builtin_amdgcn_tensor_load_to_lds) && \
    __has_builtin(__builtin_amdgcn_s_wait_tensorcnt)
#define HAVE_TDM 1
#else
#define HAVE_TDM 0
#endif

// =====================================================================
// Elementwise / utility kernels (pure bandwidth work)
// =====================================================================
__global__ void k_zero(float* __restrict__ p, long n) {
  long i = (long)blockIdx.x * blockDim.x + threadIdx.x;
  if (i < n) p[i] = 0.0f;
}

__global__ void k_f32_to_f16(const float* __restrict__ s, _Float16* __restrict__ d, long n) {
  long i = (long)blockIdx.x * blockDim.x + threadIdx.x;
  if (i < n) d[i] = (_Float16)s[i];
}

// h = 0.7*h + 0.3*ctx   (conversation_weight mix at layer 0)
__global__ void k_mix(float* __restrict__ h, const float* __restrict__ ctx, long n) {
  long i = (long)blockIdx.x * blockDim.x + threadIdx.x;
  if (i < n) h[i] = 0.7f * h[i] + 0.3f * ctx[i];
}

// histogram of an edge index list
__global__ void k_count(const int* __restrict__ idx, int nE, float* __restrict__ cnt) {
  int i = blockIdx.x * blockDim.x + threadIdx.x;
  if (i < nE) atomicAdd(&cnt[idx[i]], 1.0f);
}

// o = a + b (+ c) (+ d)
__global__ void k_add4(float* __restrict__ o, const float* __restrict__ a,
                       const float* __restrict__ b, const float* __restrict__ c,
                       const float* __restrict__ d, int n) {
  int i = blockIdx.x * blockDim.x + threadIdx.x;
  if (i < n) {
    float v = a[i] + b[i];
    if (c) v += c[i];
    if (d) v += d[i];
    o[i] = v;
  }
}

// column mean of [rows x cols] row-major matrix (avg_post)
__global__ void k_colmean(const float* __restrict__ x, int rows, int cols,
                          float* __restrict__ out) {
  __shared__ float red[256];
  int col = blockIdx.x;
  float s = 0.0f;
  for (int r = threadIdx.x; r < rows; r += blockDim.x) s += x[(long)r * cols + col];
  red[threadIdx.x] = s;
  __syncthreads();
  for (int st = 128; st > 0; st >>= 1) {
    if ((int)threadIdx.x < st) red[threadIdx.x] += red[threadIdx.x + st];
    __syncthreads();
  }
  if (threadIdx.x == 0) out[col] = red[0] / (float)rows;
}

// apc[j] = conv_b[j] + sum_k avg_post[k] * conv_W[(EFD+k)*HD + j]
// (the broadcast avg_post half of the conv-context GEMM collapsed to a vector)
__global__ void k_apc(const float* __restrict__ avgp, const float* __restrict__ convW,
                      const float* __restrict__ convb, float* __restrict__ apc) {
  int j = threadIdx.x;  // 128 threads
  float s = convb[j];
  for (int k = 0; k < IND; ++k) s += avgp[k] * convW[(EFD + k) * HD + j];
  apc[j] = s;
}

// =====================================================================
// WMMA GEMM:  out[M x 128] (+)= act( A[M x K](f16) @ B[K x 128](f16) + bias )
// block = 256 threads = 8 wave32s; block tile 64(M) x 128(N); BK = 32
// wave (wr,wc): wr in 0..3 picks 16-row strip, wc in 0..1 picks 64-col strip;
// each wave holds 4 f32 16x16 accumulators -> v_wmma_f32_16x16x32_f16.
//
// A tile staging uses the Tensor Data Mover when available: one
// tensor_load_to_lds DMA per K-step streams the 64x32 f16 tile into LDS.
// The descriptor's tensor_dim1 = M-m0 makes the TDM zero-fill ragged rows
// at the M boundary, and pad_interval=16DW/pad_amount=4DW reproduces the
// 40-half LDS row stride (32 data + 8 pad) used by the fragment loads.
// =====================================================================
#define LDSA_STRIDE 40   // halfs per A row  (80 B, 16B-aligned rows; = 32 + 8 pad)
#define LDSB_STRIDE 40   // halfs per Bt row (col-major staged B)

__launch_bounds__(256)
__global__ void k_gemm_f16_n128(const _Float16* __restrict__ A,
                                const _Float16* __restrict__ B,
                                const float* __restrict__ bias,
                                float* __restrict__ out,
                                int M, int K, int act, int accum) {
  __shared__ __align__(16) _Float16 As[64 * LDSA_STRIDE];
  __shared__ __align__(16) _Float16 Bt[128 * LDSB_STRIDE];

  const int tid  = threadIdx.x;
  const int lane = tid & 31;
  const int wave = tid >> 5;
  const int wr   = wave >> 1;   // 0..3
  const int wc   = wave & 1;    // 0..1
  const int m0   = blockIdx.x * 64;

  v8f acc[4] = {};

  const int b_kr  = tid >> 3;        // 0..31
  const int b_seg = (tid & 7) * 16;  // halfs 0..112
#if !HAVE_TDM
  const int a_row = tid >> 2;        // 0..63
  const int a_seg = (tid & 3) * 8;   // halfs 0,8,16,24
#endif
  const int ml = wr * 16 + (lane & 15);
  const int kh = lane >> 4;          // which K-half this lane owns

  for (int k0 = 0; k0 < K; k0 += 32) {
#if HAVE_TDM
    // ---- stage A tile (64 x 32 f16) via Tensor Data Mover ----
    if (wave == 0) {
      const unsigned lds_base = (unsigned)(size_t)(&As[0]);
      const unsigned long long ga =
          (unsigned long long)(size_t)(A + (long)m0 * K + k0);
      const unsigned rows = (unsigned)(M - m0);  // rows beyond M -> zero-filled
      // D# group0: count=1 | lds_addr | global_addr[56:0] | type=2
      v4u g0 = { 1u,
                 lds_base,
                 (unsigned)ga,
                 (unsigned)((ga >> 32) & 0x01FFFFFFu) | (2u << 30) };
      // D# group1: data_size=2B, pad_enable, pad_interval=16DW, pad_amount=4DW,
      //            tensor_dim0=32, tensor_dim1=rows, tile 32x64, dim0_stride=K
      v8i g1 = { (int)((1u << 16) | (1u << 20) | (3u << 22) | (3u << 25)),
                 (int)(32u << 16),                       // tensor_dim0 lo16
                 (int)((rows & 0xFFFFu) << 16),          // tensor_dim1 lo16
                 (int)((rows >> 16) | (32u << 16)),      // tensor_dim1 hi16 | tile_dim0
                 (int)64u,                               // tile_dim1=64, tile_dim2=0
                 (int)K,                                 // tensor_dim0_stride lo32
                 0, 0 };
      v4i gz4 = { 0, 0, 0, 0 };                          // 2D tensor: groups 2/3 unused
      v8i gz8 = { 0, 0, 0, 0, 0, 0, 0, 0 };
      __builtin_amdgcn_tensor_load_to_lds(g0, g1, gz4, gz4, gz8, 0);
      __builtin_amdgcn_s_wait_tensorcnt(0);
    }
#else
    // ---- fallback: stage A tile with b128 global loads ----
    {
      const int gr = m0 + a_row;
      v8h av = {};
      if (gr < M) av = *(const v8h*)(A + (long)gr * K + k0 + a_seg);
      *(v8h*)(&As[a_row * LDSA_STRIDE + a_seg]) = av;
      if (k0 + 32 < K && gr < M)
        __builtin_prefetch(A + (long)gr * K + k0 + 32 + a_seg, 0, 1);
    }
#endif

    // ---- stage B tile (32 x 128 f16) transposed into Bt[n][k] ----
    const _Float16* brow = B + (long)(k0 + b_kr) * HD + b_seg;
    v8h bv0 = *(const v8h*)(brow);
    v8h bv1 = *(const v8h*)(brow + 8);
#pragma unroll
    for (int x = 0; x < 8; ++x) Bt[(b_seg + x) * LDSB_STRIDE + b_kr] = bv0[x];
#pragma unroll
    for (int x = 0; x < 8; ++x) Bt[(b_seg + 8 + x) * LDSB_STRIDE + b_kr] = bv1[x];

    // prefetch next B tile while we compute (global_prefetch_b8)
    if (k0 + 32 < K)
      __builtin_prefetch(B + (long)(k0 + 32 + b_kr) * HD + b_seg, 0, 1);
    __syncthreads();

    // ---- A fragment: 16x32 f16 layout (lanes 0-15: K 0-7/16-23, lanes 16-31: K 8-15/24-31)
    v16h a;
    {
      const _Float16* ap = &As[ml * LDSA_STRIDE + kh * 8];
      v8h alo = *(const v8h*)(ap);
      v8h ahi = *(const v8h*)(ap + 16);
#pragma unroll
      for (int x = 0; x < 8; ++x) { a[x] = alo[x]; a[8 + x] = ahi[x]; }
    }

    // ---- 4 WMMA ops: one per 16-col subtile of this wave's 64-col strip ----
#pragma unroll
    for (int t = 0; t < 4; ++t) {
      const int n = wc * 64 + t * 16 + (lane & 15);
      const _Float16* bp = &Bt[n * LDSB_STRIDE + kh * 8];
      v8h blo = *(const v8h*)(bp);
      v8h bhi = *(const v8h*)(bp + 16);
      v16h b;
#pragma unroll
      for (int x = 0; x < 8; ++x) { b[x] = blo[x]; b[8 + x] = bhi[x]; }
      acc[t] = __builtin_amdgcn_wmma_f32_16x16x32_f16(
          /*neg_a=*/false, a, /*neg_b=*/false, b,
          /*c_mod=*/(short)0, acc[t], /*reuse_a=*/false, /*reuse_b=*/false);
    }
    __syncthreads();   // compiler inserts s_wait_dscnt before barrier: LDS reads
                       // retire before next iteration's DMA/stores overwrite tiles
  }

  // ---- epilogue: C/D layout lanes 0-15 -> M=r, lanes 16-31 -> M=8+r ----
#pragma unroll
  for (int t = 0; t < 4; ++t) {
    const int col = wc * 64 + t * 16 + (lane & 15);
#pragma unroll
    for (int r = 0; r < 8; ++r) {
      const int row = m0 + wr * 16 + (lane >> 4) * 8 + r;
      if (row < M) {
        float v = acc[t][r];
        if (bias)  v += bias[col];
        if (accum) v += out[(long)row * HD + col];
        if (act == 1)      v = v > 0.0f ? v : 0.0f;          // relu
        else if (act == 2) v = v > 0.0f ? v : 0.01f * v;     // leaky relu
        out[(long)row * HD + col] = v;
      }
    }
  }
}

// =====================================================================
// out[dst[e]] += feat[gidx[e]]   (row = 128 f32); one wave32 per edge,
// float4 gather + 4 global_atomic_add_f32 per lane.
// =====================================================================
__launch_bounds__(256)
__global__ void k_scatter_rows(const float* __restrict__ feat, const int* __restrict__ gidx,
                               const int* __restrict__ didx, int nE, float* __restrict__ out) {
  const int e = blockIdx.x * 8 + (threadIdx.x >> 5);
  if (e >= nE) return;
  const int lane = threadIdx.x & 31;
  const int g = gidx[e], d = didx[e];
  const float4 v = *(const float4*)(feat + (long)g * HD + lane * 4);
  float* o = out + (long)d * HD + lane * 4;
  atomicAdd(o + 0, v.x); atomicAdd(o + 1, v.y);
  atomicAdd(o + 2, v.z); atomicAdd(o + 3, v.w);
}

// cc row -> LayerNorm -> ReLU -> ctx[ucu_u[e]] += row    (one 128-thread block per edge)
__launch_bounds__(128)
__global__ void k_ln_relu_scatter(const float* __restrict__ cc, const int* __restrict__ uidx,
                                  const float* __restrict__ g, const float* __restrict__ beta,
                                  float* __restrict__ ctx, int nE) {
  __shared__ float red[HD];
  const int e = blockIdx.x;
  const int c = threadIdx.x;
  const float v = cc[(long)e * HD + c];
  red[c] = v; __syncthreads();
  for (int s = 64; s > 0; s >>= 1) { if (c < s) red[c] += red[c + s]; __syncthreads(); }
  const float mean = red[0] * (1.0f / HD); __syncthreads();
  const float dv = v - mean;
  red[c] = dv * dv; __syncthreads();
  for (int s = 64; s > 0; s >>= 1) { if (c < s) red[c] += red[c + s]; __syncthreads(); }
  const float var = red[0] * (1.0f / HD);
  float y = dv * rsqrtf(var + 1e-5f) * g[c] + beta[c];
  y = y > 0.0f ? y : 0.0f;
  atomicAdd(&ctx[(long)uidx[e] * HD + c], y);
}

// ctx += 0.3*sum_com/max(c_com,1) + 0.5*sum_pub/max(c_pub,1)
__global__ void k_ctx_combine(float* __restrict__ ctx, const float* __restrict__ sc,
                              const float* __restrict__ ccom, const float* __restrict__ sp,
                              const float* __restrict__ cpub, int N) {
  long i = (long)blockIdx.x * blockDim.x + threadIdx.x;
  if (i >= (long)N * HD) return;
  const int r = (int)(i >> 7);
  float dc = ccom[r]; dc = dc < 1.0f ? 1.0f : dc;
  float dp = cpub[r]; dp = dp < 1.0f ? 1.0f : dp;
  ctx[i] += 0.3f * sc[i] / dc + 0.5f * sp[i] / dp;
}

// mean-aggregate normalize fused with f16 convert (GEMM A operand)
__global__ void k_div_to_h16(const float* __restrict__ agg, const float* __restrict__ cnt,
                             _Float16* __restrict__ out, int N) {
  long i = (long)blockIdx.x * blockDim.x + threadIdx.x;
  if (i >= (long)N * HD) return;
  float d = cnt[i >> 7]; d = d < 1.0f ? 1.0f : d;
  out[i] = (_Float16)(agg[i] / d);
}

// partial sums for global-mean over nonzero-degree rows: gmacc[0..127]=col sums, gmacc[128]=row count
__launch_bounds__(128)
__global__ void k_gm_partial(const float* __restrict__ hn, const float* __restrict__ deg,
                             int N, float* __restrict__ gmacc) {
  const int c = threadIdx.x;
  float s = 0.0f, cnt = 0.0f;
  for (int r = blockIdx.x; r < N; r += gridDim.x) {
    if (deg[r] > 0.0f) { s += hn[(long)r * HD + c]; cnt += 1.0f; }
  }
  atomicAdd(&gmacc[c], s);
  if (c == 0) atomicAdd(&gmacc[HD], cnt);
}

// fix_zero_degree + LayerNorm (+ optional leaky relu); one 128-thread block per row
__launch_bounds__(128)
__global__ void k_fix_ln(const float* __restrict__ hn, const float* __restrict__ init,
                         const float* __restrict__ deg, const float* __restrict__ gmacc,
                         const float* __restrict__ g, const float* __restrict__ b,
                         float* __restrict__ out, int N, int do_lrelu) {
  __shared__ float red[HD];
  const int c = threadIdx.x;
  for (int r = blockIdx.x; r < N; r += gridDim.x) {
    const float dg = deg[r];
    float e = (dg == 0.0f) ? init[(long)r * HD + c] : hn[(long)r * HD + c];
    if (dg == 0.0f) {
      float nz = gmacc[HD]; nz = nz < 1.0f ? 1.0f : nz;
      e = 0.9f * e + 0.1f * (gmacc[c] / nz);
    }
    red[c] = e; __syncthreads();
    for (int s = 64; s > 0; s >>= 1) { if (c < s) red[c] += red[c + s]; __syncthreads(); }
    const float mean = red[0] * (1.0f / HD); __syncthreads();
    const float dv = e - mean;
    red[c] = dv * dv; __syncthreads();
    for (int s = 64; s > 0; s >>= 1) { if (c < s) red[c] += red[c + s]; __syncthreads(); }
    const float var = red[0] * (1.0f / HD); __syncthreads();
    float y = dv * rsqrtf(var + 1e-5f) * g[c] + b[c];
    if (do_lrelu) y = y > 0.0f ? y : 0.01f * y;
    out[(long)r * HD + c] = y;
  }
}

// tiny classifier tail: out[r,c] = b2[c] + sum_k h1[r,k] * W2[k,c]   (C=4)
__global__ void k_cls2(const float* __restrict__ h1, const float* __restrict__ W2,
                       const float* __restrict__ b2, float* __restrict__ out, int N) {
  long i = (long)blockIdx.x * blockDim.x + threadIdx.x;
  if (i >= (long)N * CD) return;
  const int r = (int)(i >> 2), c = (int)(i & 3);
  float s = b2[c];
  const float* hr = h1 + (long)r * HD;
  for (int k = 0; k < HD; ++k) s += hr[k] * W2[k * CD + c];
  out[i] = s;
}

// =====================================================================
// Host-side orchestration
// =====================================================================
static inline unsigned cdiv(long a, long b) { return (unsigned)((a + b - 1) / b); }

extern "C" void kernel_launch(void* const* d_in, const int* in_sizes, int n_in,
                              void* d_out, int out_size, void* d_ws, size_t ws_size,
                              hipStream_t stream) {
  // ---- inputs (setup_inputs dict order) ----
  const float* user_x      = (const float*)d_in[0];
  const float* post_x      = (const float*)d_in[1];
  const float* parent_feat = (const float*)d_in[2];
  const float* user_proj_W = (const float*)d_in[3];
  const float* user_proj_b = (const float*)d_in[4];
  const float* post_proj_W = (const float*)d_in[5];
  const float* post_proj_b = (const float*)d_in[6];
  const float* conv_W      = (const float*)d_in[7];
  const float* conv_b      = (const float*)d_in[8];
  const float* conv_g      = (const float*)d_in[9];
  const float* conv_beta   = (const float*)d_in[10];
  const float* uce_W       = (const float*)d_in[11];
  const float* uce_b       = (const float*)d_in[12];
  const float* Wself       = (const float*)d_in[13];
  const float* bself       = (const float*)d_in[14];
  const float* Wneigh      = (const float*)d_in[15];
  const float* bneigh      = (const float*)d_in[16];
  const float* ln_user_g   = (const float*)d_in[17];
  const float* ln_user_b   = (const float*)d_in[18];
  const float* ln_post_g   = (const float*)d_in[19];
  const float* ln_post_b   = (const float*)d_in[20];
  const float* cls_W1      = (const float*)d_in[21];
  const float* cls_b1      = (const float*)d_in[22];
  const float* cls_W2      = (const float*)d_in[23];
  const float* cls_b2      = (const float*)d_in[24];
  const int* publish_u     = (const int*)d_in[25];
  const int* publish_v     = (const int*)d_in[26];
  const int* comment_u     = (const int*)d_in[27];
  const int* comment_v     = (const int*)d_in[28];
  const int* ucu_u         = (const int*)d_in[29];
  const int* ucu_v         = (const int*)d_in[30];
  float* out = (float*)d_out;

  // ---- workspace bump allocator (256-B aligned) ----
  char* ws = (char*)d_ws;
  size_t off = 0;
  auto alloc = [&](size_t bytes) -> void* {
    void* p = ws + off;
    off += (bytes + 255) & ~(size_t)255;
    return p;
  };
  const size_t HH = (size_t)HD * HD;

  // f16 weights
  _Float16* wUP  = (_Float16*)alloc(2 * (size_t)IND * HD);
  _Float16* wPP  = (_Float16*)alloc(2 * (size_t)IND * HD);
  _Float16* wUCE = (_Float16*)alloc(2 * (size_t)IND * HD);
  _Float16* wCV0 = (_Float16*)alloc(2 * (size_t)EFD * HD);
  _Float16* wS   = (_Float16*)alloc(2 * (size_t)LL * 3 * HH);
  _Float16* wNg  = (_Float16*)alloc(2 * (size_t)LL * 3 * HH);
  _Float16* wC1  = (_Float16*)alloc(2 * HH);
  // f16 activation staging
  _Float16* Ah      = (_Float16*)alloc(2 * (size_t)EUc * EFD);  // fits Nu*IN too
  _Float16* hu_h    = (_Float16*)alloc(2 * (size_t)NU * HD);
  _Float16* hp_h    = (_Float16*)alloc(2 * (size_t)NPo * HD);
  _Float16* mnpub_h = (_Float16*)alloc(2 * (size_t)NPo * HD);
  _Float16* mncom_h = (_Float16*)alloc(2 * (size_t)NPo * HD);
  _Float16* mnucu_h = (_Float16*)alloc(2 * (size_t)NU * HD);
  // f32 state
  float* h_user   = (float*)alloc(4 * (size_t)NU * HD);
  float* h_post   = (float*)alloc(4 * (size_t)NPo * HD);
  float* init_u   = (float*)alloc(4 * (size_t)NU * HD);
  float* init_p   = (float*)alloc(4 * (size_t)NPo * HD);
  float* ctx      = (float*)alloc(4 * (size_t)NU * HD);
  float* post_enc = (float*)alloc(4 * (size_t)NPo * HD);
  float* cc       = (float*)alloc(4 * (size_t)EUc * HD);
  float* agg_pub  = (float*)alloc(4 * (size_t)NPo * HD);
  float* agg_com  = (float*)alloc(4 * (size_t)NPo * HD);
  float* agg_ucu  = (float*)alloc(4 * (size_t)NU * HD);
  float* user_new = (float*)alloc(4 * (size_t)NU * HD);
  float* post_new = (float*)alloc(4 * (size_t)NPo * HD);
  float* sum_com  = agg_ucu;   // reuse: free until layer loop
  float* sum_pub  = user_new;  // reuse: free until layer loop
  // counts / small vectors
  float* c_pub_u  = (float*)alloc(4 * (size_t)NU);
  float* c_com_u  = (float*)alloc(4 * (size_t)NU);
  float* c_ucu_u  = (float*)alloc(4 * (size_t)NU);
  float* c_ucu_v  = (float*)alloc(4 * (size_t)NU);
  float* c_pub_v  = (float*)alloc(4 * (size_t)NPo);
  float* c_com_v  = (float*)alloc(4 * (size_t)NPo);
  float* deg_user = (float*)alloc(4 * (size_t)NU);
  float* deg_post = (float*)alloc(4 * (size_t)NPo);
  float* avgp     = (float*)alloc(4 * (size_t)IND);
  float* apcv     = (float*)alloc(4 * (size_t)HD);
  float* gm_u     = (float*)alloc(4 * (size_t)(HD + 1));
  float* gm_p     = (float*)alloc(4 * (size_t)(HD + 1));
  (void)ws_size; (void)in_sizes; (void)n_in; (void)out_size;

  const long NuH = (long)NU * HD, NpH = (long)NPo * HD;
  #define Z(p, n)      k_zero<<<cdiv((n), 256), 256, 0, stream>>>((p), (long)(n))
  #define TOH(s, d, n) k_f32_to_f16<<<cdiv((n), 256), 256, 0, stream>>>((s), (d), (long)(n))
  #define GEMM(A_, B_, bias_, O_, M_, K_, act_, acc_) \
    k_gemm_f16_n128<<<cdiv((M_), 64), 256, 0, stream>>>((A_), (B_), (bias_), (O_), (M_), (K_), (act_), (acc_))
  #define SCAT(f, g, d, n, o) k_scatter_rows<<<cdiv((n), 8), 256, 0, stream>>>((f), (g), (d), (n), (o))

  // ---- 0: zero all accumulation state (ws is not re-poisoned between replays) ----
  Z(c_pub_u, NU); Z(c_com_u, NU); Z(c_ucu_u, NU); Z(c_ucu_v, NU);
  Z(c_pub_v, NPo); Z(c_com_v, NPo);
  Z(ctx, NuH);

  // ---- 1: degree / segment counts ----
  k_count<<<cdiv(EPu, 256), 256, 0, stream>>>(publish_u, EPu, c_pub_u);
  k_count<<<cdiv(ECo, 256), 256, 0, stream>>>(comment_u, ECo, c_com_u);
  k_count<<<cdiv(EUc, 256), 256, 0, stream>>>(ucu_u, EUc, c_ucu_u);
  k_count<<<cdiv(EUc, 256), 256, 0, stream>>>(ucu_v, EUc, c_ucu_v);
  k_count<<<cdiv(EPu, 256), 256, 0, stream>>>(publish_v, EPu, c_pub_v);
  k_count<<<cdiv(ECo, 256), 256, 0, stream>>>(comment_v, ECo, c_com_v);
  k_add4<<<cdiv(NU, 256), 256, 0, stream>>>(deg_user, c_pub_u, c_com_u, c_ucu_u, c_ucu_v, NU);
  k_add4<<<cdiv(NPo, 256), 256, 0, stream>>>(deg_post, c_pub_v, c_com_v, nullptr, nullptr, NPo);

  // ---- 2: weights -> f16 (conv_W first 64 rows are the parent_feat part, contiguous) ----
  TOH(user_proj_W, wUP,  (long)IND * HD);
  TOH(post_proj_W, wPP,  (long)IND * HD);
  TOH(uce_W,       wUCE, (long)IND * HD);
  TOH(conv_W,      wCV0, (long)EFD * HD);
  TOH(Wself,       wS,   (long)LL * 3 * HH);
  TOH(Wneigh,      wNg,  (long)LL * 3 * HH);
  TOH(cls_W1,      wC1,  (long)HH);

  // ---- 3: input projections (WMMA) ----
  TOH(user_x, Ah, (long)NU * IND);
  GEMM(Ah, wUP, user_proj_b, h_user, NU, IND, 2, 0);     // lrelu
  TOH(post_x, Ah, (long)NPo * IND);
  GEMM(Ah, wPP, post_proj_b, h_post, NPo, IND, 2, 0);    // lrelu
  GEMM(Ah, wUCE, uce_b, post_enc, NPo, IND, 1, 0);       // relu (same staged post_x)
  (void)hipMemcpyAsync(init_u, h_user, 4 * (size_t)NuH, hipMemcpyDeviceToDevice, stream);
  (void)hipMemcpyAsync(init_p, h_post, 4 * (size_t)NpH, hipMemcpyDeviceToDevice, stream);

  // ---- 4: conversation context ----
  k_colmean<<<IND, 256, 0, stream>>>(post_x, NPo, IND, avgp);
  k_apc<<<1, HD, 0, stream>>>(avgp, conv_W, conv_b, apcv);
  TOH(parent_feat, Ah, (long)EUc * EFD);
  GEMM(Ah, wCV0, apcv, cc, EUc, EFD, 0, 0);              // pre-LN linear, bias = apc vector
  k_ln_relu_scatter<<<EUc, HD, 0, stream>>>(cc, ucu_u, conv_g, conv_beta, ctx, EUc);
  Z(sum_com, NuH); Z(sum_pub, NuH);
  SCAT(post_enc, comment_v, comment_u, ECo, sum_com);
  SCAT(post_enc, publish_v, publish_u, EPu, sum_pub);
  k_ctx_combine<<<cdiv(NuH, 256), 256, 0, stream>>>(ctx, sum_com, c_com_u, sum_pub, c_pub_u, NU);

  // ---- 5: SAGE layers ----
  for (int i = 0; i < LL; ++i) {
    if (i == 0) k_mix<<<cdiv(NuH, 256), 256, 0, stream>>>(h_user, ctx, NuH);

    Z(agg_pub, NpH); Z(agg_com, NpH); Z(agg_ucu, NuH);
    SCAT(h_user, publish_u, publish_v, EPu, agg_pub);
    SCAT(h_user, comment_u, comment_v, ECo, agg_com);
    SCAT(h_user, ucu_u, ucu_v, EUc, agg_ucu);

    TOH(h_user, hu_h, NuH);
    TOH(h_post, hp_h, NpH);
    k_div_to_h16<<<cdiv(NpH, 256), 256, 0, stream>>>(agg_pub, c_pub_v, mnpub_h, NPo);
    k_div_to_h16<<<cdiv(NpH, 256), 256, 0, stream>>>(agg_com, c_com_v, mncom_h, NPo);
    k_div_to_h16<<<cdiv(NuH, 256), 256, 0, stream>>>(agg_ucu, c_ucu_v, mnucu_h, NU);

    // post_new = hP@Ws0+bs0 + mnPub@Wn0+bn0 + hP@Ws1+bs1 + mnCom@Wn1+bn1
    GEMM(hp_h,    wS  + (size_t)(i * 3 + 0) * HH, bself  + (i * 3 + 0) * HD, post_new, NPo, HD, 0, 0);
    GEMM(mnpub_h, wNg + (size_t)(i * 3 + 0) * HH, bneigh + (i * 3 + 0) * HD, post_new, NPo, HD, 0, 1);
    GEMM(hp_h,    wS  + (size_t)(i * 3 + 1) * HH, bself  + (i * 3 + 1) * HD, post_new, NPo, HD, 0, 1);
    GEMM(mncom_h, wNg + (size_t)(i * 3 + 1) * HH, bneigh + (i * 3 + 1) * HD, post_new, NPo, HD, 0, 1);
    // user_new = hU@Ws2+bs2 + mnUcu@Wn2+bn2
    GEMM(hu_h,    wS  + (size_t)(i * 3 + 2) * HH, bself  + (i * 3 + 2) * HD, user_new, NU, HD, 0, 0);
    GEMM(mnucu_h, wNg + (size_t)(i * 3 + 2) * HH, bneigh + (i * 3 + 2) * HD, user_new, NU, HD, 0, 1);

    // zero-degree fixup (global mean over nonzero-degree rows) + LN (+lrelu if not last)
    Z(gm_u, HD + 1); Z(gm_p, HD + 1);
    k_gm_partial<<<1024, HD, 0, stream>>>(user_new, deg_user, NU, gm_u);
    k_gm_partial<<<1024, HD, 0, stream>>>(post_new, deg_post, NPo, gm_p);
    k_fix_ln<<<NU, HD, 0, stream>>>(user_new, init_u, deg_user, gm_u, ln_user_g, ln_user_b,
                                    h_user, NU, i < LL - 1);
    k_fix_ln<<<NPo, HD, 0, stream>>>(post_new, init_p, deg_post, gm_p, ln_post_g, ln_post_b,
                                     h_post, NPo, i < LL - 1);
  }

  // ---- 6: classifier head ----
  TOH(h_user, hu_h, NuH);
  GEMM(hu_h, wC1, cls_b1, user_new /*reuse as hidden*/, NU, HD, 2, 0);  // lrelu
  k_cls2<<<cdiv((long)NU * CD, 256), 256, 0, stream>>>(user_new, cls_W2, cls_b2, out, NU);

  #undef Z
  #undef TOH
  #undef GEMM
  #undef SCAT
}
